// SpikingDenseLayer_11493332484485
// MI455X (gfx1250) — matrix operand order repaired and checked by reading, not compiled, not run on hardware
//
#include <hip/hip_runtime.h>

typedef __attribute__((ext_vector_type(16))) __bf16 v16bf;
typedef __attribute__((ext_vector_type(8)))  float  v8f;

namespace {
constexpr int kB = 256, kT = 100, kF = 1024, kU = 1024;
constexpr int BT = 16;               // batch rows per block (WMMA M)
constexpr int UT = 64;               // u columns per block (4 waves x 16)
constexpr int NWAVE = UT / 16;
constexpr int NTHREADS = NWAVE * 32; // 128 threads = 4 wave32
constexpr int KP = kF / 2;           // 512 bf16 pairs along K
constexpr int KLDS_DW = KP * UT;     // 32768 dwords = 128 KB (weight tile)
constexpr int XLDS_DW = KP * BT;     // 8192 dwords  =  32 KB (one x tile)
constexpr float TAU = 0.25f;
}

__device__ __forceinline__ unsigned int pack2_bf16(float a, float b) {
  union { float f; unsigned int u; } ca, cb;
  ca.f = a; cb.f = b;
  unsigned int x = ca.u, y = cb.u;
  x = (x + 0x7FFFu + ((x >> 16) & 1u)) >> 16;   // round-to-nearest-even
  y = (y + 0x7FFFu + ((y >> 16) & 1u)) >> 16;
  return (x & 0xFFFFu) | (y << 16);
}

extern __shared__ unsigned int s_lds[];

__global__ __launch_bounds__(NTHREADS, 1)
void snn_dense_lif(const float* __restrict__ x,
                   const float* __restrict__ w,
                   const float* __restrict__ bias,
                   float* __restrict__ out_spikes,
                   float* __restrict__ out_counts) {
  unsigned int* klds  = s_lds;                       // [KP][UT] bf16 pairs of W
  unsigned int* xlds0 = s_lds + KLDS_DW;             // [KP][BT] x tile, timestep t
  unsigned int* xlds1 = s_lds + KLDS_DW + XLDS_DW;   // [KP][BT] x tile, timestep t+1

  const int b0 = blockIdx.x * BT;
  const int u0 = blockIdx.y * UT;
  const int tid  = threadIdx.x;
  const int wave = tid >> 5;
  const int lane = tid & 31;

  // ---- stage weight tile once: klds[kp*UT + uu] = pack(w[2kp][u], w[2kp+1][u])
  for (int p = tid; p < KLDS_DW; p += NTHREADS) {
    const int kp = p / UT;
    const int uu = p % UT;
    const float lo = w[(size_t)(2 * kp)     * kU + (u0 + uu)];
    const float hi = w[(size_t)(2 * kp + 1) * kU + (u0 + uu)];
    klds[p] = pack2_bf16(lo, hi);
  }

  const int n    = lane & 15;
  const int ksel = lane >> 4;          // 0: low K half, 1: high K half (ISA layout)
  const int mb   = ksel << 3;          // C/D row offset: lanes 16-31 hold M+8
  const int ucol = u0 + wave * 16 + n;
  const float bv = bias[ucol];

  v8f volt, cnt;
  #pragma unroll
  for (int r = 0; r < 8; ++r) { volt[r] = 0.0f; cnt[r] = 0.0f; }

  __syncthreads();

  // ---- two timesteps per pass: one B-fragment load feeds two WMMAs
  for (int t = 0; t < kT; t += 2) {
    for (int p = tid; p < XLDS_DW; p += NTHREADS) {
      const int m  = p >> 9;           // p / KP
      const int fp = p & (KP - 1);     // feature-pair index
      const size_t base = ((size_t)(b0 + m) * kT + t) * kF + 2 * fp;
      const float2 v0 = *(const float2*)&x[base];
      const float2 v1 = *(const float2*)&x[base + kF];   // same b, t+1
      xlds0[fp * BT + m] = pack2_bf16(v0.x, v0.y);
      xlds1[fp * BT + m] = pack2_bf16(v1.x, v1.y);
    }
    __syncthreads();

    v8f acc0, acc1;
    #pragma unroll
    for (int r = 0; r < 8; ++r) { acc0[r] = 0.0f; acc1[r] = 0.0f; }

    #pragma unroll 4
    for (int kk = 0; kk < kF / 32; ++kk) {
      const int kb2 = kk * 16;         // pair-base of this 32-wide K chunk
      union { unsigned int u[8]; v16bf v; } af0, af1, bf;
      #pragma unroll
      for (int j = 0; j < 8; ++j) {
        const int jj = j + (j & 4);    // j=0..3 -> 0..3 ; j=4..7 -> 8..11
        // A (16x32 bf16): lane<16 K∈{0..7,16..23}, lane>=16 K∈{8..15,24..31}
        const int axy = (kb2 + ksel * 4 + jj) * BT + n;
        af0.u[j] = xlds0[axy];
        af1.u[j] = xlds1[axy];
        // B (32x16 bf16): lane<16 K∈{0..7,16..23}, lane>=16 K∈{8..15,24..31}, N=lane%16
        bf.u[j] = klds[(kb2 + ksel * 4 + jj) * UT + (wave * 16 + n)];
      }
      acc0 = __builtin_amdgcn_wmma_f32_16x16x32_bf16(
          false, af0.v, false, bf.v, (short)0, acc0, false, false);
      acc1 = __builtin_amdgcn_wmma_f32_16x16x32_bf16(
          false, af1.v, false, bf.v, (short)0, acc1, false, false);
    }

    // ---- fused LIF update for both timesteps (state stays in VGPRs)
    #pragma unroll
    for (int r = 0; r < 8; ++r) {
      const int brow = b0 + mb + r;
      const size_t orow = ((size_t)brow * kT + t) * kU + ucol;

      const float cur0 = acc0[r] + bv;
      float v = TAU * volt[r] + (1.0f - TAU) * cur0;
      const float sp0 = (v > 1.0f) ? 1.0f : 0.0f;
      v -= sp0;                        // VTH == 1.0
      out_spikes[orow] = sp0;

      const float cur1 = acc1[r] + bv;
      v = TAU * v + (1.0f - TAU) * cur1;
      const float sp1 = (v > 1.0f) ? 1.0f : 0.0f;
      v -= sp1;
      out_spikes[orow + kU] = sp1;

      volt[r] = v;
      cnt[r] += sp0 + sp1;
    }
    __syncthreads();                   // x tiles are overwritten next pass
  }

  #pragma unroll
  for (int r = 0; r < 8; ++r) {
    const int brow = b0 + mb + r;
    out_counts[(size_t)brow * kU + ucol] = cnt[r];
  }
}

extern "C" void kernel_launch(void* const* d_in, const int* in_sizes, int n_in,
                              void* d_out, int out_size, void* d_ws, size_t ws_size,
                              hipStream_t stream) {
  const float* x    = (const float*)d_in[0];
  const float* w    = (const float*)d_in[1];
  const float* bias = (const float*)d_in[2];
  float* spikes = (float*)d_out;
  float* counts = spikes + (size_t)kB * kT * kU;

  dim3 grid(kB / BT, kU / UT);         // 16 x 16 = 256 blocks
  dim3 block(NTHREADS);                // 128 threads = 4 wave32
  const size_t lds_bytes = (size_t)(KLDS_DW + 2 * XLDS_DW) * 4;  // 192 KB of 320 KB
  snn_dense_lif<<<grid, block, lds_bytes, stream>>>(x, w, bias, spikes, counts);
}
